// CrossCoder_25761213841690
// MI455X (gfx1250) — matrix operand order repaired and checked by reading, not compiled, not run on hardware
//
#include <hip/hip_runtime.h>

// ---------------------------------------------------------------------------
// CrossCoder forward for MI455X (gfx1250, wave32, WMMA + async-LDS):
//   0) cvt kernels   : f32 -> bf16 copies of x and selected W_enc (if ws fits)
//   1) encode kernel : h = x @ W_enc^T + b_enc  (v_wmma_f32_16x16x32_bf16,
//                      double-buffered LDS tiles via GLOBAL_LOAD_ASYNC_TO_LDS)
//   2) topk_kernel   : exact per-row top-128 (radix select on float bits)
//   3) decode_kernel : out = sum_k s_k * W_dec[idx_k,:] + b_dec (f32 gather)
// ---------------------------------------------------------------------------

static constexpr int BB = 4096;   // batch
static constexpr int DD = 2048;   // model dim
static constexpr int HH = 16384;  // latent dim
static constexpr int KK = 128;    // top-k

typedef __attribute__((ext_vector_type(16))) __bf16          v16bf;
typedef __attribute__((ext_vector_type(8)))  float           v8f;
typedef __attribute__((ext_vector_type(8)))  unsigned short  v8us;
typedef __attribute__((ext_vector_type(16))) unsigned short  v16us;

__device__ __forceinline__ unsigned short f32_to_bf16_bits(float f) {
    unsigned u = __builtin_bit_cast(unsigned, f);
    unsigned r = u + 0x7FFFu + ((u >> 16) & 1u);   // round-to-nearest-even
    return (unsigned short)(r >> 16);
}

// -------- gfx1250 async global->LDS copy (signature confirmed by probe) -----
#if __has_builtin(__builtin_amdgcn_global_load_async_to_lds_b128)
  #define HAVE_ASYNC_LDS 1
  typedef int v4i_raw __attribute__((vector_size(4 * sizeof(int))));
  typedef __attribute__((address_space(1))) v4i_raw global_v4i;
  typedef __attribute__((address_space(3))) v4i_raw lds_v4i;
#else
  #define HAVE_ASYNC_LDS 0
#endif

__device__ __forceinline__ void wait_async_and_barrier() {
#if HAVE_ASYNC_LDS
  #if __has_builtin(__builtin_amdgcn_s_wait_asynccnt)
    __builtin_amdgcn_s_wait_asynccnt(0);
  #else
    asm volatile("s_wait_asynccnt 0x0" ::: "memory");
  #endif
#endif
    __syncthreads();
}

// copy 32 consecutive f32 (128B) global -> LDS for this thread
__device__ __forceinline__ void stage_f32_row32(const float* __restrict__ g, float* l) {
#if HAVE_ASYNC_LDS
    #pragma unroll
    for (int c = 0; c < 8; c++)
        __builtin_amdgcn_global_load_async_to_lds_b128(
            (global_v4i*)(g + 4 * c), (lds_v4i*)(l + 4 * c), 0, 0);
#else
    float4 t[8];
    #pragma unroll
    for (int c = 0; c < 8; c++) t[c] = ((const float4*)g)[c];
    #pragma unroll
    for (int c = 0; c < 8; c++) ((float4*)l)[c] = t[c];
#endif
}

// copy 32 consecutive bf16 (64B) global -> LDS for this thread
__device__ __forceinline__ void stage_bf16_row32(const unsigned short* __restrict__ g,
                                                 unsigned short* l) {
#if HAVE_ASYNC_LDS
    #pragma unroll
    for (int c = 0; c < 4; c++)
        __builtin_amdgcn_global_load_async_to_lds_b128(
            (global_v4i*)(g + 8 * c), (lds_v4i*)(l + 8 * c), 0, 0);
#else
    uint4 t[4];
    #pragma unroll
    for (int c = 0; c < 4; c++) t[c] = ((const uint4*)g)[c];
    #pragma unroll
    for (int c = 0; c < 4; c++) ((uint4*)l)[c] = t[c];
#endif
}

// ---------------------------------------------------------------------------
// f32 -> bf16 conversion pre-passes (bandwidth bound, ~80MB total)
// ---------------------------------------------------------------------------
__global__ __launch_bounds__(256) void cvt_x_kernel(
    const float* __restrict__ src, unsigned short* __restrict__ dst)
{
    const long long i = ((long long)blockIdx.x * 256 + threadIdx.x) * 4;
    float4 v = *(const float4*)(src + i);
    union { unsigned long long u; unsigned short e[4]; } p;
    p.e[0] = f32_to_bf16_bits(v.x); p.e[1] = f32_to_bf16_bits(v.y);
    p.e[2] = f32_to_bf16_bits(v.z); p.e[3] = f32_to_bf16_bits(v.w);
    *(unsigned long long*)(dst + i) = p.u;
}

__global__ __launch_bounds__(256) void cvt_w_kernel(
    const float* __restrict__ W0, const float* __restrict__ W1,
    const int* __restrict__ em, unsigned short* __restrict__ dst)
{
    const float* __restrict__ src = (*em == 0) ? W0 : W1;
    const long long i = ((long long)blockIdx.x * 256 + threadIdx.x) * 4;
    float4 v = *(const float4*)(src + i);
    union { unsigned long long u; unsigned short e[4]; } p;
    p.e[0] = f32_to_bf16_bits(v.x); p.e[1] = f32_to_bf16_bits(v.y);
    p.e[2] = f32_to_bf16_bits(v.z); p.e[3] = f32_to_bf16_bits(v.w);
    *(unsigned long long*)(dst + i) = p.u;
}

// ---------------------------------------------------------------------------
// Shared GEMM geometry: 128x128 tile per WG, 256 threads (8 waves, 2x4), TK=64
// ---------------------------------------------------------------------------
static constexpr int TM = 128, TN = 128, TK = 64;

// ======================= bf16-staged encode (primary) =======================
static constexpr int SAh = 72;   // LDS row stride (bf16 elems): 144B, 16B-aligned chunks

__global__ __launch_bounds__(256) void encode_bf16_kernel(
    const unsigned short* __restrict__ xb,   // [BB][DD] bf16 bits
    const unsigned short* __restrict__ wb,   // [HH][DD] bf16 bits
    const float* __restrict__ be0, const float* __restrict__ be1,
    const int*   __restrict__ em,
    float* __restrict__ h)
{
    const float* __restrict__ be = (*em == 0) ? be0 : be1;

    __shared__ unsigned short As[2][TM * SAh];
    __shared__ unsigned short Bs[2][TN * SAh];

    const int tid  = threadIdx.x;
    const int lane = tid & 31;
    const int wave = tid >> 5;
    const int wm   = wave & 1;
    const int wn   = wave >> 1;

    const int bM = blockIdx.y * TM;
    const int bN = blockIdx.x * TN;

    const int srow = tid >> 1;
    const int scol = (tid & 1) * 32;          // 32 bf16 = 64B per thread

    const int  l16 = lane & 15;
    const bool hiL = (lane >= 16);

    const unsigned short* gx = xb + (size_t)(bM + srow) * DD + scol;
    const unsigned short* gw = wb + (size_t)(bN + srow) * DD + scol;
    const int lofs = srow * SAh + scol;

    v8f acc[4][2];
    #pragma unroll
    for (int ms = 0; ms < 4; ms++)
        #pragma unroll
        for (int ns = 0; ns < 2; ns++)
            #pragma unroll
            for (int r = 0; r < 8; r++)
                acc[ms][ns][r] = 0.0f;

    constexpr int nIter = DD / TK;   // 32

    stage_bf16_row32(gx, &As[0][lofs]);
    stage_bf16_row32(gw, &Bs[0][lofs]);
    wait_async_and_barrier();

    for (int it = 0; it < nIter; it++) {
        const int cur = it & 1;

        if (it + 1 < nIter) {
            stage_bf16_row32(gx + (it + 1) * TK, &As[cur ^ 1][lofs]);
            stage_bf16_row32(gw + (it + 1) * TK, &Bs[cur ^ 1][lofs]);
        }
        if (it + 2 < nIter) {                 // global_prefetch_b8 warm-up
            __builtin_prefetch(gx + (it + 2) * TK, 0, 3);
            __builtin_prefetch(gw + (it + 2) * TK, 0, 3);
        }

        #pragma unroll
        for (int ks = 0; ks < 2; ks++) {
            const int kb = 32 * ks;

            // A 16x32: lanes 0-15 hold K{0-7,16-23}; lanes 16-31 K{8-15,24-31}
            v16bf af[4];
            #pragma unroll
            for (int ms = 0; ms < 4; ms++) {
                const unsigned short* p =
                    &As[cur][(wm * 64 + ms * 16 + l16) * SAh + kb + (hiL ? 8 : 0)];
                union { v16us u; v8us half[2]; } f;
                f.half[0] = *(const v8us*)p;
                f.half[1] = *(const v8us*)(p + 16);
                af[ms] = __builtin_bit_cast(v16bf, f.u);
            }

            // B 32x16: lane holds column N=l16; K{0-7,16-23} (lo lanes) / K{8-15,24-31} (hi lanes)
            v16bf bfr[2];
            #pragma unroll
            for (int ns = 0; ns < 2; ns++) {
                const unsigned short* p =
                    &Bs[cur][(wn * 32 + ns * 16 + l16) * SAh + kb + (hiL ? 8 : 0)];
                union { v16us u; v8us half[2]; } f;
                f.half[0] = *(const v8us*)p;
                f.half[1] = *(const v8us*)(p + 16);
                bfr[ns] = __builtin_bit_cast(v16bf, f.u);
            }

            #pragma unroll
            for (int ms = 0; ms < 4; ms++)
                #pragma unroll
                for (int ns = 0; ns < 2; ns++)
                    acc[ms][ns] = __builtin_amdgcn_wmma_f32_16x16x32_bf16(
                        false, af[ms], false, bfr[ns], (short)0,
                        acc[ms][ns], false, false);
        }

        wait_async_and_barrier();
    }

    const int mh = hiL ? 8 : 0;
    #pragma unroll
    for (int ns = 0; ns < 2; ns++) {
        const int col = bN + wn * 32 + ns * 16 + l16;
        const float bias = be[col];
        #pragma unroll
        for (int ms = 0; ms < 4; ms++) {
            const size_t base = (size_t)(bM + wm * 64 + ms * 16 + mh) * HH + col;
            #pragma unroll
            for (int r = 0; r < 8; r++)
                h[base + (size_t)r * HH] = acc[ms][ns][r] + bias;
        }
    }
}

// ============== f32-staged encode (fallback when ws is tight) ===============
static constexpr int SAf = 68;   // LDS row stride (floats)

__global__ __launch_bounds__(256) void encode_f32_kernel(
    const float* __restrict__ x,
    const float* __restrict__ We0, const float* __restrict__ be0,
    const float* __restrict__ We1, const float* __restrict__ be1,
    const int*   __restrict__ em,
    float* __restrict__ h)
{
    const int sel = *em;
    const float* __restrict__ We = (sel == 0) ? We0 : We1;
    const float* __restrict__ be = (sel == 0) ? be0 : be1;

    __shared__ float As[2][TM * SAf];
    __shared__ float Bs[2][TN * SAf];

    const int tid  = threadIdx.x;
    const int lane = tid & 31;
    const int wave = tid >> 5;
    const int wm   = wave & 1;
    const int wn   = wave >> 1;

    const int bM = blockIdx.y * TM;
    const int bN = blockIdx.x * TN;

    const int srow = tid >> 1;
    const int scol = (tid & 1) * 32;

    const int  l16 = lane & 15;
    const bool hiL = (lane >= 16);

    const float* gx = x  + (size_t)(bM + srow) * DD + scol;
    const float* gw = We + (size_t)(bN + srow) * DD + scol;
    const int lofs = srow * SAf + scol;

    v8f acc[4][2];
    #pragma unroll
    for (int ms = 0; ms < 4; ms++)
        #pragma unroll
        for (int ns = 0; ns < 2; ns++)
            #pragma unroll
            for (int r = 0; r < 8; r++)
                acc[ms][ns][r] = 0.0f;

    constexpr int nIter = DD / TK;

    stage_f32_row32(gx, &As[0][lofs]);
    stage_f32_row32(gw, &Bs[0][lofs]);
    wait_async_and_barrier();

    for (int it = 0; it < nIter; it++) {
        const int cur = it & 1;

        if (it + 1 < nIter) {
            stage_f32_row32(gx + (it + 1) * TK, &As[cur ^ 1][lofs]);
            stage_f32_row32(gw + (it + 1) * TK, &Bs[cur ^ 1][lofs]);
        }
        if (it + 2 < nIter) {
            __builtin_prefetch(gx + (it + 2) * TK, 0, 3);
            __builtin_prefetch(gw + (it + 2) * TK, 0, 3);
        }

        #pragma unroll
        for (int ks = 0; ks < 2; ks++) {
            const int kb = 32 * ks;

            v16bf af[4];
            #pragma unroll
            for (int ms = 0; ms < 4; ms++) {
                const float* p =
                    &As[cur][(wm * 64 + ms * 16 + l16) * SAf + kb + (hiL ? 8 : 0)];
                float4 c0 = ((const float4*)p)[0];
                float4 c1 = ((const float4*)p)[1];
                float4 c2 = *(const float4*)(p + 16);
                float4 c3 = *(const float4*)(p + 20);
                union { v16us u; unsigned short e[16]; } f;
                f.e[0]  = f32_to_bf16_bits(c0.x); f.e[1]  = f32_to_bf16_bits(c0.y);
                f.e[2]  = f32_to_bf16_bits(c0.z); f.e[3]  = f32_to_bf16_bits(c0.w);
                f.e[4]  = f32_to_bf16_bits(c1.x); f.e[5]  = f32_to_bf16_bits(c1.y);
                f.e[6]  = f32_to_bf16_bits(c1.z); f.e[7]  = f32_to_bf16_bits(c1.w);
                f.e[8]  = f32_to_bf16_bits(c2.x); f.e[9]  = f32_to_bf16_bits(c2.y);
                f.e[10] = f32_to_bf16_bits(c2.z); f.e[11] = f32_to_bf16_bits(c2.w);
                f.e[12] = f32_to_bf16_bits(c3.x); f.e[13] = f32_to_bf16_bits(c3.y);
                f.e[14] = f32_to_bf16_bits(c3.z); f.e[15] = f32_to_bf16_bits(c3.w);
                af[ms] = __builtin_bit_cast(v16bf, f.u);
            }

            v16bf bfr[2];
            #pragma unroll
            for (int ns = 0; ns < 2; ns++) {
                const float* p =
                    &Bs[cur][(wn * 32 + ns * 16 + l16) * SAf + kb + (hiL ? 8 : 0)];
                float4 c0 = ((const float4*)p)[0];
                float4 c1 = ((const float4*)p)[1];
                float4 c2 = *(const float4*)(p + 16);
                float4 c3 = *(const float4*)(p + 20);
                union { v16us u; unsigned short e[16]; } f;
                f.e[0]  = f32_to_bf16_bits(c0.x); f.e[1]  = f32_to_bf16_bits(c0.y);
                f.e[2]  = f32_to_bf16_bits(c0.z); f.e[3]  = f32_to_bf16_bits(c0.w);
                f.e[4]  = f32_to_bf16_bits(c1.x); f.e[5]  = f32_to_bf16_bits(c1.y);
                f.e[6]  = f32_to_bf16_bits(c1.z); f.e[7]  = f32_to_bf16_bits(c1.w);
                f.e[8]  = f32_to_bf16_bits(c2.x); f.e[9]  = f32_to_bf16_bits(c2.y);
                f.e[10] = f32_to_bf16_bits(c2.z); f.e[11] = f32_to_bf16_bits(c2.w);
                f.e[12] = f32_to_bf16_bits(c3.x); f.e[13] = f32_to_bf16_bits(c3.y);
                f.e[14] = f32_to_bf16_bits(c3.z); f.e[15] = f32_to_bf16_bits(c3.w);
                bfr[ns] = __builtin_bit_cast(v16bf, f.u);
            }

            #pragma unroll
            for (int ms = 0; ms < 4; ms++)
                #pragma unroll
                for (int ns = 0; ns < 2; ns++)
                    acc[ms][ns] = __builtin_amdgcn_wmma_f32_16x16x32_bf16(
                        false, af[ms], false, bfr[ns], (short)0,
                        acc[ms][ns], false, false);
        }

        wait_async_and_barrier();
    }

    const int mh = hiL ? 8 : 0;
    #pragma unroll
    for (int ns = 0; ns < 2; ns++) {
        const int col = bN + wn * 32 + ns * 16 + l16;
        const float bias = be[col];
        #pragma unroll
        for (int ms = 0; ms < 4; ms++) {
            const size_t base = (size_t)(bM + wm * 64 + ms * 16 + mh) * HH + col;
            #pragma unroll
            for (int r = 0; r < 8; r++)
                h[base + (size_t)r * HH] = acc[ms][ns][r] + bias;
        }
    }
}

// ---------------------------------------------------------------------------
// Exact top-128 per row: 4-round radix select on order-preserving uint bits,
// deterministic index-ordered compaction (lowest-index tie-break).
// ---------------------------------------------------------------------------
__device__ __forceinline__ unsigned mapf(float f) {
    unsigned b = __builtin_bit_cast(unsigned, f);
    return (b & 0x80000000u) ? ~b : (b | 0x80000000u);
}
__device__ __forceinline__ float unmapf(unsigned v) {
    unsigned b = (v & 0x80000000u) ? (v ^ 0x80000000u) : ~v;
    return __builtin_bit_cast(float, b);
}

__global__ __launch_bounds__(512) void topk_kernel(
    const float* __restrict__ h,
    float* __restrict__ vals, int* __restrict__ idxs)
{
    __shared__ unsigned hist[256];
    __shared__ int scanG[513];
    __shared__ int scanE[512];
    __shared__ unsigned sh_prefix;
    __shared__ int sh_kRem;

    const int b = blockIdx.x;
    const int t = threadIdx.x;
    const float* __restrict__ row = h + (size_t)b * HH;

    unsigned u[32];
    #pragma unroll
    for (int i = 0; i < 32; i += 4) {
        float4 v4 = *(const float4*)(row + t * 32 + i);
        u[i + 0] = mapf(v4.x); u[i + 1] = mapf(v4.y);
        u[i + 2] = mapf(v4.z); u[i + 3] = mapf(v4.w);
    }

    if (t == 0) { sh_prefix = 0u; sh_kRem = KK; }
    unsigned pmask = 0u;

    for (int shift = 24; shift >= 0; shift -= 8) {
        if (t < 256) hist[t] = 0u;
        __syncthreads();
        const unsigned prefix = sh_prefix;
        #pragma unroll
        for (int i = 0; i < 32; i++)
            if ((u[i] & pmask) == prefix)
                atomicAdd(&hist[(u[i] >> shift) & 255u], 1u);
        __syncthreads();
        if (t == 0) {
            int kRem = sh_kRem, cum = 0, selb = 0;
            for (int v = 255; v >= 0; v--) {
                int c = (int)hist[v];
                if (cum + c >= kRem) { selb = v; break; }
                cum += c;
            }
            sh_kRem   = kRem - cum;
            sh_prefix = prefix | ((unsigned)selb << shift);
        }
        pmask |= (0xFFu << shift);
        __syncthreads();
    }

    const unsigned T = sh_prefix;
    const int needEq = sh_kRem;

    int cg = 0, ce = 0;
    #pragma unroll
    for (int i = 0; i < 32; i++) { cg += (u[i] > T); ce += (u[i] == T); }
    scanG[t] = cg; scanE[t] = ce;
    __syncthreads();
    if (t == 0) {
        int aG = 0, aE = 0;
        for (int i = 0; i < 512; i++) {
            int g = scanG[i], e = scanE[i];
            scanG[i] = aG; scanE[i] = aE;
            aG += g; aE += e;
        }
        scanG[512] = aG;
    }
    __syncthreads();

    int offG = scanG[t], offE = scanE[t];
    const int G = scanG[512];
    float* __restrict__ vrow = vals + (size_t)b * KK;
    int*   __restrict__ irow = idxs + (size_t)b * KK;
    #pragma unroll
    for (int i = 0; i < 32; i++) {
        const int gidx = t * 32 + i;
        if (u[i] > T) {
            const int pos = offG++;
            vrow[pos] = unmapf(u[i]); irow[pos] = gidx;
        } else if (u[i] == T) {
            const int e = offE++;
            if (e < needEq) {
                const int pos = G + e;
                vrow[pos] = unmapf(u[i]); irow[pos] = gidx;
            }
        }
    }
}

// ---------------------------------------------------------------------------
// Sparse decode: out[b,:] = b_dec + sum_j vals[b,j] * W_dec[idx[b,j],:]
// ---------------------------------------------------------------------------
__global__ __launch_bounds__(512) void decode_kernel(
    const float* __restrict__ Wd0, const float* __restrict__ bd0,
    const float* __restrict__ Wd1, const float* __restrict__ bd1,
    const int*   __restrict__ dm,
    const float* __restrict__ vals, const int* __restrict__ idxs,
    float* __restrict__ out)
{
    const int sel = *dm;
    const float* __restrict__ Wd = (sel == 0) ? Wd0 : Wd1;
    const float* __restrict__ bd = (sel == 0) ? bd0 : bd1;

    __shared__ float sv[KK];
    __shared__ int   si[KK];

    const int b = blockIdx.x;
    const int t = threadIdx.x;
    if (t < KK) {
        sv[t] = vals[(size_t)b * KK + t];
        si[t] = idxs[(size_t)b * KK + t];
    }
    __syncthreads();

    const int d0 = t * 4;
    float4 acc = *(const float4*)(bd + d0);
    #pragma unroll 4
    for (int j = 0; j < KK; j++) {
        const float  v = sv[j];
        const float4 w = *(const float4*)(Wd + (size_t)si[j] * DD + d0);
        acc.x += v * w.x; acc.y += v * w.y;
        acc.z += v * w.z; acc.w += v * w.w;
    }
    *(float4*)(out + (size_t)b * DD + d0) = acc;
}

// ---------------------------------------------------------------------------
extern "C" void kernel_launch(void* const* d_in, const int* in_sizes, int n_in,
                              void* d_out, int out_size, void* d_ws, size_t ws_size,
                              hipStream_t stream) {
    (void)in_sizes; (void)n_in; (void)out_size;
    const float* x   = (const float*)d_in[0];
    const float* We0 = (const float*)d_in[1];
    const float* be0 = (const float*)d_in[2];
    const float* We1 = (const float*)d_in[3];
    const float* be1 = (const float*)d_in[4];
    const float* Wd0 = (const float*)d_in[5];
    const float* bd0 = (const float*)d_in[6];
    const float* Wd1 = (const float*)d_in[7];
    const float* bd1 = (const float*)d_in[8];
    const int*   em  = (const int*)d_in[9];
    const int*   dm  = (const int*)d_in[10];
    float* out = (float*)d_out;

    // workspace layout
    float* h    = (float*)d_ws;                           // [BB,HH] fp32
    float* vals = h + (size_t)BB * HH;                    // [BB,KK] fp32
    int*   idxs = (int*)(vals + (size_t)BB * KK);         // [BB,KK] i32
    unsigned short* xb = (unsigned short*)(idxs + (size_t)BB * KK);  // [BB,DD] bf16
    unsigned short* wb = xb + (size_t)BB * DD;                       // [HH,DD] bf16

    const size_t need_base = (size_t)BB * HH * 4 + (size_t)BB * KK * 8;
    const size_t need_bf16 = need_base + ((size_t)BB * DD + (size_t)HH * DD) * 2;

    dim3 ge(HH / TN, BB / TM);                            // 128 x 32 tiles

    if (ws_size >= need_bf16) {
        // pre-convert x and selected W_enc to bf16, then bf16-staged WMMA GEMM
        cvt_x_kernel<<<(int)(((size_t)BB * DD) / 1024), 256, 0, stream>>>(x, xb);
        cvt_w_kernel<<<(int)(((size_t)HH * DD) / 1024), 256, 0, stream>>>(We0, We1, em, wb);
        encode_bf16_kernel<<<ge, 256, 0, stream>>>(xb, wb, be0, be1, em, h);
    } else {
        encode_f32_kernel<<<ge, 256, 0, stream>>>(x, We0, be0, We1, be1, em, h);
    }
    topk_kernel<<<BB, 512, 0, stream>>>(h, vals, idxs);
    decode_kernel<<<BB, 512, 0, stream>>>(Wd0, bd0, Wd1, bd1, dm, vals, idxs, out);
}
